// MultiheadSelfAttention_87960930222761
// MI455X (gfx1250) — matrix operand order repaired and checked by reading, not compiled, not run on hardware
//
#include <hip/hip_runtime.h>

// ---------------------------------------------------------------------------
// Multi-head causal self-attention for MI455X (gfx1250, wave32, WMMA).
// B=2, S=2048, D=1024, H=16, dh=64.
//   k0: fp32 -> bf16 conversion of x / qkv_w / o_w into workspace
//   k1: QKV GEMM (bf16 WMMA, async global->LDS double-buffered) -> Q(/8), K, V^T
//   k2: flash attention (bf16 WMMA, online softmax) -> attn bf16 workspace
//   k3: output GEMM (same WMMA pipeline) -> fp32 d_out
// ---------------------------------------------------------------------------

typedef __attribute__((ext_vector_type(16))) __bf16         v16bf;
typedef __attribute__((ext_vector_type(8)))  float          v8f;
typedef __attribute__((ext_vector_type(4)))  float          f32x4;
typedef __attribute__((ext_vector_type(4)))  unsigned int   u32x4;
typedef __attribute__((ext_vector_type(4)))  int            i32x4;
typedef __attribute__((ext_vector_type(4)))  unsigned short u16x4;

#define DMODEL 1024
#define SEQ    2048
#define NHEAD  16
#define DHEAD  64

#define GLOBAL_AS __attribute__((address_space(1)))
#define LDS_AS    __attribute__((address_space(3)))

// gfx1250 async global->LDS copy (ASYNCcnt-tracked). Guarded: falls back to a
// sync VGPR round-trip if this toolchain lacks the builtin. The builtin is
// typed on the 128-bit payload: (global int4*, lds int4*, imm offset, imm cpol).
#if defined(__AMDGCN__) && __has_builtin(__builtin_amdgcn_global_load_async_to_lds_b128)
#define USE_ASYNC_CP 1
#else
#define USE_ASYNC_CP 0
#endif

__device__ __forceinline__ void async_cp_b128(const void* g, void* l) {
#if USE_ASYNC_CP
    __builtin_amdgcn_global_load_async_to_lds_b128(
        (GLOBAL_AS i32x4*)g, (LDS_AS i32x4*)l, 0, 0);
#else
    *(u32x4*)l = *(const u32x4*)g;
#endif
}

__device__ __forceinline__ void wait_async0() {
#if defined(__AMDGCN__) && __has_builtin(__builtin_amdgcn_s_wait_asynccnt)
    __builtin_amdgcn_s_wait_asynccnt(0);
#elif defined(__AMDGCN__)
    asm volatile("s_wait_asynccnt 0x0" ::: "memory");
#endif
}

union Frag16 {
    struct { u32x4 lo, hi; } u;   // two 16B chunks = 32 bytes = one bf16 WMMA operand
    v16bf v;
};

__device__ __forceinline__ unsigned short f2bf_bits(float x) {
    unsigned int u = __builtin_bit_cast(unsigned int, x);
    u += 0x7FFFu + ((u >> 16) & 1u);            // round-to-nearest-even
    return (unsigned short)(u >> 16);
}
__device__ __forceinline__ __bf16 f2bf(float x) {
    unsigned short h = f2bf_bits(x);
    return __builtin_bit_cast(__bf16, h);
}
__device__ __forceinline__ v8f zero8() {
    v8f z;
#pragma unroll
    for (int i = 0; i < 8; ++i) z[i] = 0.0f;
    return z;
}
__device__ __forceinline__ v8f wmma_bf16(v16bf a, v16bf b, v8f c) {
    // D = A(16x32 bf16) * B(32x16 bf16) + C(16x16 f32)
    return __builtin_amdgcn_wmma_f32_16x16x32_bf16(false, a, false, b,
                                                   (short)0, c, false, false);
}

// ---------------------------------------------------------------------------
// k0: elementwise fp32 -> bf16 (float4 in, 4x bf16 out)
// ---------------------------------------------------------------------------
__global__ void __launch_bounds__(256)
cvt_bf16_kernel(const float* __restrict__ src, unsigned short* __restrict__ dst, int n4) {
    int i = blockIdx.x * 256 + threadIdx.x;
    if (i >= n4) return;
    f32x4 v = ((const f32x4*)src)[i];
    u16x4 o;
    o.x = f2bf_bits(v.x); o.y = f2bf_bits(v.y);
    o.z = f2bf_bits(v.z); o.w = f2bf_bits(v.w);
    ((u16x4*)dst)[i] = o;
}

// ---------------------------------------------------------------------------
// GEMM: C[M x N] = A[M x 1024] * B^T, A/B bf16 row-major with K=1024 contiguous.
// Block = 256 threads (8 waves), tile 128(M) x 128(N), K-step 32,
// double-buffered LDS fed by async global->LDS copies (tile t+1 in flight
// while tile t computes). 80B LDS row stride -> conflict-free ds_load_b128.
// Wave w: wm=w&3 -> 32-row strip, wn=w>>2 -> 64-col strip: 2x4 WMMA accs.
// mode 0: scatter QKV epilogue (bf16, Q pre-scaled 1/8, V transposed)
// mode 1: fp32 store to out
// ---------------------------------------------------------------------------
__global__ void __launch_bounds__(256)
gemm_bf16_kernel(const unsigned short* __restrict__ Aus,
                 const unsigned short* __restrict__ Bus,
                 float* __restrict__ outf,
                 unsigned short* __restrict__ Qus,
                 unsigned short* __restrict__ Kus,
                 unsigned short* __restrict__ Vtus,
                 int mode) {
    __shared__ __align__(16) unsigned int As[2][128 * 20];   // 2 x 10KB
    __shared__ __align__(16) unsigned int Bs[2][128 * 20];

    const __bf16* A = (const __bf16*)Aus;
    const __bf16* B = (const __bf16*)Bus;

    const int tid  = threadIdx.x;
    const int lane = tid & 31;
    const int w    = tid >> 5;
    const int half = lane >> 4;
    const int l16  = lane & 15;
    const int wm   = w & 3;     // M strip: 32*wm
    const int wn   = w >> 2;    // N strip: 64*wn
    const int Mbase = blockIdx.y * 128;
    const int Nbase = blockIdx.x * 128;

    // Each thread stages 2x16B of A and 2x16B of B per k-step (512 chunks/tile).
    auto stage = [&](int ks, int buf) {
#pragma unroll
        for (int i = 0; i < 2; ++i) {
            int c   = tid + i * 256;
            int row = c >> 2;
            int col = c & 3;                        // 16B chunk within row
            const __bf16* ga = A + (size_t)(Mbase + row) * 1024 + ks * 32 + col * 8;
            const __bf16* gb = B + (size_t)(Nbase + row) * 1024 + ks * 32 + col * 8;
            async_cp_b128(ga, &As[buf][row * 20 + col * 4]);
            async_cp_b128(gb, &Bs[buf][row * 20 + col * 4]);
        }
    };

    v8f acc[2][4];
#pragma unroll
    for (int mt = 0; mt < 2; ++mt)
#pragma unroll
        for (int nt = 0; nt < 4; ++nt) acc[mt][nt] = zero8();

    const int NK = 1024 / 32;
    stage(0, 0);
    wait_async0();
    __syncthreads();

    for (int kt = 0; kt < NK; ++kt) {
        const int cur = kt & 1;
        if (kt + 1 < NK) stage(kt + 1, cur ^ 1);   // async fill of other buffer

        // Gather all 6 fragments first so ds_loads batch ahead of the WMMAs.
        // A frag (16x32): lane<16 row=l16 K[0..15]; lane>=16 K[16..31]
        Frag16 af[2];
#pragma unroll
        for (int mt = 0; mt < 2; ++mt) {
            int r = wm * 32 + mt * 16 + l16;
            af[mt].u.lo = *(const u32x4*)(&As[cur][r * 20 + half * 8]);
            af[mt].u.hi = *(const u32x4*)(&As[cur][r * 20 + half * 8 + 4]);
        }
        // B frag (32x16): lane<16 col=l16 K[0..15]; lane>=16 K[16..31]
        Frag16 bfr[4];
#pragma unroll
        for (int nt = 0; nt < 4; ++nt) {
            int rn = wn * 64 + nt * 16 + l16;
            bfr[nt].u.lo = *(const u32x4*)(&Bs[cur][rn * 20 + half * 8]);
            bfr[nt].u.hi = *(const u32x4*)(&Bs[cur][rn * 20 + half * 8 + 4]);
        }
#pragma unroll
        for (int nt = 0; nt < 4; ++nt)
#pragma unroll
            for (int mt = 0; mt < 2; ++mt)
                acc[mt][nt] = wmma_bf16(af[mt].v, bfr[nt].v, acc[mt][nt]);

        wait_async0();       // next tile landed (this wave's copies)
        __syncthreads();     // all waves' copies visible; all reads of old buf done
    }

    // Epilogue: C element (M = r + 8*half, N = l16) per VGPR r.
    __bf16* Qp  = (__bf16*)Qus;
    __bf16* Kp  = (__bf16*)Kus;
    __bf16* Vtp = (__bf16*)Vtus;
#pragma unroll
    for (int mt = 0; mt < 2; ++mt)
#pragma unroll
        for (int nt = 0; nt < 4; ++nt)
#pragma unroll
            for (int r = 0; r < 8; ++r) {
                int grow = Mbase + wm * 32 + mt * 16 + r + 8 * half;   // 0..4095
                int gcol = Nbase + wn * 64 + nt * 16 + l16;
                float v = acc[mt][nt][r];
                if (mode == 1) {
                    outf[(size_t)grow * 1024 + gcol] = v;
                } else {
                    int b = grow >> 11, s = grow & 2047;
                    int proj = gcol >> 10, hh = (gcol >> 6) & 15, d = gcol & 63;
                    size_t bh = (size_t)(b * NHEAD + hh);
                    if (proj == 0)       Qp[(bh * SEQ + s) * DHEAD + d] = f2bf(v * 0.125f);
                    else if (proj == 1)  Kp[(bh * SEQ + s) * DHEAD + d] = f2bf(v);
                    else                 Vtp[(bh * DHEAD + d) * SEQ + s] = f2bf(v);
                }
            }
}

// ---------------------------------------------------------------------------
// k2: causal flash attention. Grid (S/128, B*H), 256 threads = 8 waves.
// Each wave owns 16 query rows, streams key blocks of 32 with online softmax.
// Q fragments live in registers; K/V B-fragments loaded straight from global
// (per-(b,h) K+V is 512KB -> resident in 192MB L2); V loads hoisted above the
// softmax so they overlap the exp/shuffle VALU work; P transposed C-layout ->
// A-layout through a per-wave LDS buffer (no barriers needed).
// ---------------------------------------------------------------------------
__global__ void __launch_bounds__(256)
fa_kernel(const unsigned short* __restrict__ Qus,
          const unsigned short* __restrict__ Kus,
          const unsigned short* __restrict__ Vtus,
          unsigned short* __restrict__ attnus) {
    __shared__ __align__(16) unsigned short Plds[8 * 16 * 40];  // per-wave 16x32, 80B stride

    const __bf16* Q  = (const __bf16*)Qus;
    const __bf16* K  = (const __bf16*)Kus;
    const __bf16* Vt = (const __bf16*)Vtus;
    __bf16* attn     = (__bf16*)attnus;

    const int tid  = threadIdx.x;
    const int lane = tid & 31;
    const int w    = tid >> 5;
    const int half = lane >> 4;
    const int l16  = lane & 15;
    const int bh   = blockIdx.y;                 // b*16 + h
    const int qw   = blockIdx.x * 128 + w * 16;  // wave's first query row

    const __bf16* Qh = Q  + (size_t)bh * SEQ * DHEAD;
    const __bf16* Kh = K  + (size_t)bh * SEQ * DHEAD;
    const __bf16* Vh = Vt + (size_t)bh * DHEAD * SEQ;

    // Q A-fragments for dh = [0,32) and [32,64): kept in registers all kernel.
    Frag16 qf[2];
#pragma unroll
    for (int kf = 0; kf < 2; ++kf) {
        const __bf16* p = Qh + (size_t)(qw + l16) * DHEAD + kf * 32 + half * 16;
        qf[kf].u.lo = *(const u32x4*)p;
        qf[kf].u.hi = *(const u32x4*)(p + 8);
    }

    float mrow[8], lrow[8];
    v8f o[4];
#pragma unroll
    for (int r = 0; r < 8; ++r) { mrow[r] = -3.0e38f; lrow[r] = 0.0f; }
#pragma unroll
    for (int nt = 0; nt < 4; ++nt) o[nt] = zero8();

    const int kend = qw + 15;             // last key any of this wave's rows can see
    const int nkb  = (kend >> 5) + 1;     // key blocks of 32

    for (int kb = 0; kb < nkb; ++kb) {
        const int k0 = kb * 32;

        // ---- scores S = (Q/8) K^T over 32 keys: 2 N-tiles x 2 K-steps ----
        v8f s0 = zero8(), s1 = zero8();
#pragma unroll
        for (int kt = 0; kt < 2; ++kt) {
            Frag16 b0, b1;
            {
                const __bf16* p = Kh + (size_t)(k0 + l16) * DHEAD + kt * 32 + half * 16;
                b0.u.lo = *(const u32x4*)p;  b0.u.hi = *(const u32x4*)(p + 8);
            }
            {
                const __bf16* p = Kh + (size_t)(k0 + 16 + l16) * DHEAD + kt * 32 + half * 16;
                b1.u.lo = *(const u32x4*)p;  b1.u.hi = *(const u32x4*)(p + 8);
            }
            s0 = wmma_bf16(qf[kt].v, b0.v, s0);
            s1 = wmma_bf16(qf[kt].v, b1.v, s1);
        }

        // ---- hoist V B-fragments: global loads overlap softmax VALU work ----
        Frag16 vf[4];
#pragma unroll
        for (int nt = 0; nt < 4; ++nt) {
            const __bf16* p = Vh + (size_t)(nt * 16 + l16) * SEQ + k0 + half * 16;
            vf[nt].u.lo = *(const u32x4*)p;  vf[nt].u.hi = *(const u32x4*)(p + 8);
        }

        // ---- causal mask (only near the diagonal) ----
        if (k0 + 31 > qw) {
#pragma unroll
            for (int r = 0; r < 8; ++r) {
                int q  = qw + r + 8 * half;
                int j0 = k0 + l16;
                if (j0 > q)      s0[r] = -3.0e38f;
                if (j0 + 16 > q) s1[r] = -3.0e38f;
            }
        }

        // ---- online softmax: rows live across the 16 lanes of each half ----
        float bm[8];
#pragma unroll
        for (int r = 0; r < 8; ++r) bm[r] = fmaxf(s0[r], s1[r]);
#pragma unroll
        for (int m = 1; m < 16; m <<= 1)
#pragma unroll
            for (int r = 0; r < 8; ++r) bm[r] = fmaxf(bm[r], __shfl_xor(bm[r], m, 32));

        float scale[8], rs[8];
#pragma unroll
        for (int r = 0; r < 8; ++r) {
            float mn = fmaxf(mrow[r], bm[r]);
            scale[r] = __expf(mrow[r] - mn);
            mrow[r]  = mn;
        }
#pragma unroll
        for (int r = 0; r < 8; ++r) {
            s0[r] = __expf(s0[r] - mrow[r]);
            s1[r] = __expf(s1[r] - mrow[r]);
            rs[r] = s0[r] + s1[r];
        }
#pragma unroll
        for (int m = 1; m < 16; m <<= 1)
#pragma unroll
            for (int r = 0; r < 8; ++r) rs[r] += __shfl_xor(rs[r], m, 32);
#pragma unroll
        for (int r = 0; r < 8; ++r) lrow[r] = lrow[r] * scale[r] + rs[r];
#pragma unroll
        for (int nt = 0; nt < 4; ++nt)
#pragma unroll
            for (int r = 0; r < 8; ++r) o[nt][r] *= scale[r];

        // ---- P: C-layout -> A-layout via per-wave LDS (no barrier needed) ----
        unsigned short* pb = &Plds[w * 640];
#pragma unroll
        for (int r = 0; r < 8; ++r) {
            int row = r + 8 * half;
            pb[row * 40 + l16]      = f2bf_bits(s0[r]);
            pb[row * 40 + 16 + l16] = f2bf_bits(s1[r]);
        }
        Frag16 pf;
        pf.u.lo = *(const u32x4*)(&pb[l16 * 40 + half * 16]);
        pf.u.hi = *(const u32x4*)(&pb[l16 * 40 + half * 16 + 8]);

        // ---- O += P * V  (V^T rows give 32B contiguous key runs per dh) ----
#pragma unroll
        for (int nt = 0; nt < 4; ++nt)
            o[nt] = wmma_bf16(pf.v, vf[nt].v, o[nt]);
    }

    // ---- finalize: O /= l, store bf16 to attn[b, s, h*64 + d] ----
    const int b = bh >> 4, h = bh & 15;
#pragma unroll
    for (int nt = 0; nt < 4; ++nt)
#pragma unroll
        for (int r = 0; r < 8; ++r) {
            int q = qw + r + 8 * half;
            float v = o[nt][r] / lrow[r];
            attn[((size_t)(b * SEQ + q)) * DMODEL + h * DHEAD + nt * 16 + l16] = f2bf(v);
        }
}

// ---------------------------------------------------------------------------
// Launcher. Workspace layout (bytes):
//   [0,8M)   x bf16        [8M,14M) qkv_w bf16   [14M,16M) o_w bf16
//   [16M,24M) Q bf16       [24M,32M) K bf16      [32M,40M) V^T bf16
//   [40M,48M) attn bf16
// ---------------------------------------------------------------------------
extern "C" void kernel_launch(void* const* d_in, const int* in_sizes, int n_in,
                              void* d_out, int out_size, void* d_ws, size_t ws_size,
                              hipStream_t stream) {
    const float* x     = (const float*)d_in[0];
    const float* qkv_w = (const float*)d_in[1];
    const float* o_w   = (const float*)d_in[2];
    float* out = (float*)d_out;

    char* ws = (char*)d_ws;
    const size_t MB = 1024 * 1024;
    unsigned short* xb  = (unsigned short*)(ws + 0 * MB);
    unsigned short* wq  = (unsigned short*)(ws + 8 * MB);
    unsigned short* wo  = (unsigned short*)(ws + 14 * MB);
    unsigned short* Qp  = (unsigned short*)(ws + 16 * MB);
    unsigned short* Kp  = (unsigned short*)(ws + 24 * MB);
    unsigned short* Vtp = (unsigned short*)(ws + 32 * MB);
    unsigned short* att = (unsigned short*)(ws + 40 * MB);

    // k0: fp32 -> bf16
    cvt_bf16_kernel<<<4096, 256, 0, stream>>>(x,     xb, (2 * SEQ * DMODEL) / 4);
    cvt_bf16_kernel<<<3072, 256, 0, stream>>>(qkv_w, wq, (3 * DMODEL * DMODEL) / 4);
    cvt_bf16_kernel<<<1024, 256, 0, stream>>>(o_w,   wo, (DMODEL * DMODEL) / 4);

    // k1: QKV projection, C[4096 x 3072] -> Q, K, V^T
    gemm_bf16_kernel<<<dim3(3072 / 128, 4096 / 128), 256, 0, stream>>>(
        xb, wq, out, Qp, Kp, Vtp, /*mode=*/0);

    // k2: flash attention -> attn[4096 x 1024] bf16
    fa_kernel<<<dim3(SEQ / 128, 2 * NHEAD), 256, 0, stream>>>(Qp, Kp, Vtp, att);

    // k3: output projection, C[4096 x 1024] fp32 -> d_out
    gemm_bf16_kernel<<<dim3(1024 / 128, 4096 / 128), 256, 0, stream>>>(
        att, wo, out, Qp, Kp, Vtp, /*mode=*/1);
}
